// Group_Conv_84731114815961
// MI455X (gfx1250) — compile-verified, matrix operand order checked by reading, not run on hardware
//
#include <hip/hip_runtime.h>

typedef __attribute__((ext_vector_type(16))) __bf16  v16bf;
typedef __attribute__((ext_vector_type(8)))  float   v8f;

#define NBATCH 16
#define INCH   64
#define HH     128
#define WW     128
#define OUTCH  128
#define SUBCH  8
#define OH     126
#define OW     126
#define KTOT   576   // K ordered (ky,kx,c), c fastest: k = (ky*3+kx)*64 + c
#define NKT    18    // 576 / 32

__device__ __forceinline__ unsigned short f32_bf16(float f) {
  unsigned int u = __float_as_uint(f);
  u += 0x7FFFu + ((u >> 16) & 1u);          // round-to-nearest-even
  return (unsigned short)(u >> 16);
}

__device__ __forceinline__ v16bf pack16(uint4 lo, uint4 hi) {
  union { uint4 q[2]; v16bf v; } u;
  u.q[0] = lo; u.q[1] = hi;
  return u.v;
}

// Build dense bf16 weight matrix Wd[o][(ky*3+kx)*64 + c]:
//   s = (c - o) mod 64;  Wd = W[o,s,ky,kx] if s < 8 else 0
__global__ void prep_wd(const float* __restrict__ W, unsigned short* __restrict__ Wd) {
  int idx = blockIdx.x * 256 + threadIdx.x;   // 128*576 = 73728 total
  int o  = idx / KTOT;
  int k  = idx - o * KTOT;
  int kk = k >> 6;                            // ky*3 + kx
  int c  = k & 63;
  int s  = (c - o + 256) & 63;
  float v = (s < SUBCH) ? W[(o * SUBCH + s) * 9 + kk] : 0.0f;
  Wd[idx] = f32_bf16(v);
}

// One block per (batch, output row). 128 threads = 4 waves.
// Wave wv computes output channels [32*wv, 32*wv+32) as two 16x16 WMMA M-tiles.
__global__ __launch_bounds__(128)
void conv_wmma(const float* __restrict__ X, const unsigned short* __restrict__ Wd,
               const float* __restrict__ bias, float* __restrict__ Out) {
  // LDS: bf16 X tile, layout [ky][x(130)][c(64)], channel index XOR-swizzled by x
  __shared__ __align__(16) unsigned short Xs[3 * 130 * 64];

  const int b    = blockIdx.x / OH;
  const int y    = blockIdx.x - b * OH;
  const int tid  = threadIdx.x;
  const int lane = tid & 31;
  const int wv   = tid >> 5;          // 0..3
  const int m    = lane & 15;         // M row (A) / N col (B,C)
  const int half = lane >> 4;         // lane group 0/1
  const int o0   = wv << 5;           // first of two 16-channel tiles

  // ---- stage X rows y..y+2 into LDS (fp32 -> bf16), coalesced over x ----
  for (int i = tid; i < INCH * 3 * WW; i += 128) {
    int x   = i & (WW - 1);
    int row = i >> 7;
    int ky  = row % 3;
    int c   = row / 3;
    float v = X[((b * INCH + c) * HH + (y + ky)) * WW + x];
    Xs[(ky * 130 + x) * 64 + (c ^ ((x & 7) << 3))] = f32_bf16(v);
  }
  __syncthreads();

  // ---- persistent A fragments for both M-tiles (L2-resident Wd) ----
  v16bf a0[NKT], a1[NKT];
  {
    const unsigned short* r0 = Wd + (o0 + m) * KTOT + half * 8;  // 16B aligned
    const unsigned short* r1 = r0 + 16 * KTOT;
    #pragma unroll
    for (int kt = 0; kt < NKT; ++kt) {
      const uint4* p0 = (const uint4*)(r0 + kt * 32);
      const uint4* p1 = (const uint4*)(r1 + kt * 32);
      a0[kt] = pack16(p0[0], p0[2]);   // K = half*8+e  and  K = 16+half*8+e
      a1[kt] = pack16(p1[0], p1[2]);
    }
  }

  // ---- bias into accumulator-init fragments (C layout: M = r + 8*half, N = m) ----
  v8f cinit0, cinit1;
  #pragma unroll
  for (int r = 0; r < 8; ++r) {
    cinit0[r] = bias[o0 + half * 8 + r];
    cinit1[r] = bias[o0 + 16 + half * 8 + r];
  }

  // ---- loop over 8 pixel tiles of 16 columns ----
  for (int xt = 0; xt < 8; ++xt) {
    const int x0 = xt * 16;
    v8f acc0 = cinit0, acc1 = cinit1;
    #pragma unroll
    for (int kt = 0; kt < NKT; ++kt) {
      const int ky = kt / 6;
      const int kx = (kt >> 1) % 3;
      const int cb = ((kt & 1) << 5) + (half << 3);   // multiple of 8
      const int xx = x0 + m + kx;                      // <= 129, within padded row
      const int sw = (xx & 7) << 3;
      const unsigned short* rowp = Xs + (ky * 130 + xx) * 64;
      uint4 lo = *(const uint4*)(rowp + (cb ^ sw));         // c = cb..cb+7
      uint4 hi = *(const uint4*)(rowp + ((cb + 16) ^ sw));  // c = cb+16..cb+23
      v16bf bf = pack16(lo, hi);
      acc0 = __builtin_amdgcn_wmma_f32_16x16x32_bf16(
                 false, a0[kt], false, bf, (short)0, acc0, false, false);
      acc1 = __builtin_amdgcn_wmma_f32_16x16x32_bf16(
                 false, a1[kt], false, bf, (short)0, acc1, false, false);
    }
    if (x0 + m < OW) {
      #pragma unroll
      for (int r = 0; r < 8; ++r) {
        int oA = o0 + half * 8 + r;
        Out[((b * OUTCH + oA) * OH + y) * OW + x0 + m]        = acc0[r];
        Out[((b * OUTCH + oA + 16) * OH + y) * OW + x0 + m]   = acc1[r];
      }
    }
  }
}

extern "C" void kernel_launch(void* const* d_in, const int* in_sizes, int n_in,
                              void* d_out, int out_size, void* d_ws, size_t ws_size,
                              hipStream_t stream) {
  (void)in_sizes; (void)n_in; (void)out_size; (void)ws_size;
  const float* X    = (const float*)d_in[0];
  const float* W    = (const float*)d_in[1];
  const float* bias = (const float*)d_in[2];
  float* Out        = (float*)d_out;
  unsigned short* Wd = (unsigned short*)d_ws;   // 128*576 bf16 = 147456 bytes

  prep_wd<<<dim3((OUTCH * KTOT) / 256), dim3(256), 0, stream>>>(W, Wd);
  conv_wmma<<<dim3(NBATCH * OH), dim3(128), 0, stream>>>(X, Wd, bias, Out);
}
